// MAB_70987219468669
// MI455X (gfx1250) — matrix operand order repaired and checked by reading, not compiled, not run on hardware
//
#include <hip/hip_runtime.h>
#include <hip/hip_bf16.h>
#include <math.h>

// ---------- CDNA5 WMMA types ----------
typedef __attribute__((ext_vector_type(16))) __bf16 bf16x16;
typedef __attribute__((ext_vector_type(8)))  float  floatx8;

union Frag { uint4 q[2]; bf16x16 v; };

__device__ __forceinline__ floatx8 wmma_bf16(const Frag& a, const Frag& b, floatx8 c) {
    // D = A(16x32 bf16) * B(32x16 bf16) + C(16x16 f32)
    return __builtin_amdgcn_wmma_f32_16x16x32_bf16(false, a.v, false, b.v,
                                                   (short)0, c, false, false);
}

// Native bf16 conversion (clang selects hw bf16 cvt on gfx1250)
__device__ __forceinline__ unsigned short f2bf(float f) {
    union { __bf16 h; unsigned short u; } x;
    x.h = (__bf16)f;
    return x.u;
}
__device__ __forceinline__ unsigned pack2(float lo, float hi) {
    return (unsigned)f2bf(lo) | ((unsigned)f2bf(hi) << 16);
}
__device__ __forceinline__ float bf2f(unsigned short u) {
    union { unsigned u; float f; } x; x.u = ((unsigned)u) << 16;
    return x.f;
}

// dims
#define BATCH 8
#define NSEQ  1024
#define DIM   512
#define HEADS 8
#define DHEAD 64

// ============================================================================
// Kernel 0: stream-convert f32 activations (Q, K) to bf16 once.
// grid = (2048, 2); 8 elems/thread, b128 in / b128 out.
// ============================================================================
__global__ __launch_bounds__(256)
void cvt_bf16_kernel(const float* __restrict__ Qin, const float* __restrict__ Kin,
                     unsigned short* __restrict__ Qb, unsigned short* __restrict__ Kb)
{
    const float* src = blockIdx.y ? Kin : Qin;
    unsigned short* dst = blockIdx.y ? Kb : Qb;
    size_t i = ((size_t)blockIdx.x * 256 + threadIdx.x) * 8;
    float4 f0 = *(const float4*)(src + i);
    float4 f1 = *(const float4*)(src + i + 4);
    uint4 o = make_uint4(pack2(f0.x, f0.y), pack2(f0.z, f0.w),
                         pack2(f1.x, f1.y), pack2(f1.z, f1.w));
    *(uint4*)(dst + i) = o;
}

// ============================================================================
// Generic WMMA GEMM core: block = 256 (8 waves), workgroup tile 256x32.
// Wave w: 32 rows x 32 cols = 4 accumulator tiles (2 A-frags x 2 B-frags ->
// 4 WMMAs per K-step; 8 b128 loads per 4 WMMAs). Entire 512x32 W panel
// staged once to LDS (transposed bf16, 32 KB); inner K-loop is barrier-free.
// wldsT layout: [n (0..31)][k (0..511)], row stride 512 u16 (1 KB).
// acc[0]=A0B0, acc[1]=A0B1, acc[2]=A1B0, acc[3]=A1B1.
// ============================================================================
__device__ __forceinline__ void gemm_core(const unsigned short* __restrict__ Abf,
                                          const float* __restrict__ W,
                                          unsigned short* wldsT,
                                          int m0, int n0, int lane,
                                          floatx8 acc[4])
{
    const int half = lane >> 4;
    const int l16  = lane & 15;
    const int ac = half ? 8 : 0;    // 16-bit A-layout K sub-offset
    const int bc = half ? 8 : 0;    // 16-bit B-layout K sub-offset

    // stage W[0..511][n0..n0+31] transposed -> bf16 (16 passes, float4 each)
    {
        const int kk0 = threadIdx.x >> 3;        // 0..31
        const int nn4 = (threadIdx.x & 7) * 4;   // 0..28
        #pragma unroll
        for (int p = 0; p < 16; ++p) {
            int kk = kk0 + p * 32;
            float4 w = *(const float4*)(W + (size_t)kk * DIM + n0 + nn4);
            wldsT[(nn4 + 0) * DIM + kk] = f2bf(w.x);
            wldsT[(nn4 + 1) * DIM + kk] = f2bf(w.y);
            wldsT[(nn4 + 2) * DIM + kk] = f2bf(w.z);
            wldsT[(nn4 + 3) * DIM + kk] = f2bf(w.w);
        }
    }
    __syncthreads();

    const unsigned short* arow0 = Abf + (size_t)(m0 + l16) * DIM;
    const unsigned short* arow1 = Abf + (size_t)(m0 + 16 + l16) * DIM;
    #pragma unroll 4
    for (int k0 = 0; k0 < DIM; k0 += 32) {
        Frag a0, a1, b0, b1;
        a0.q[0] = *(const uint4*)(arow0 + k0 + ac);
        a0.q[1] = *(const uint4*)(arow0 + k0 + ac + 16);
        a1.q[0] = *(const uint4*)(arow1 + k0 + ac);
        a1.q[1] = *(const uint4*)(arow1 + k0 + ac + 16);
        const unsigned short* wp0 = &wldsT[l16 * DIM + k0 + bc];
        const unsigned short* wp1 = &wldsT[(16 + l16) * DIM + k0 + bc];
        b0.q[0] = *(const uint4*)(wp0);
        b0.q[1] = *(const uint4*)(wp0 + 16);
        b1.q[0] = *(const uint4*)(wp1);
        b1.q[1] = *(const uint4*)(wp1 + 16);
        acc[0] = wmma_bf16(a0, b0, acc[0]);
        acc[1] = wmma_bf16(a0, b1, acc[1]);
        acc[2] = wmma_bf16(a1, b0, acc[2]);
        acc[3] = wmma_bf16(a1, b1, acc[3]);
    }
}

// ============================================================================
// Kernel 1: QKV projections. grid = (32, 16, 3). A = bf16 activations.
// Output permuted [B,H,N,64] bf16 with bias.
// ============================================================================
__global__ __launch_bounds__(256)
void qkv_proj_kernel(const unsigned short* __restrict__ Qb,
                     const unsigned short* __restrict__ Kb,
                     const float* __restrict__ Wq, const float* __restrict__ bq,
                     const float* __restrict__ Wk, const float* __restrict__ bk,
                     const float* __restrict__ Wv, const float* __restrict__ bvp,
                     unsigned short* __restrict__ qbf,
                     unsigned short* __restrict__ kbf,
                     unsigned short* __restrict__ vbf)
{
    const int z = blockIdx.z;
    const unsigned short* A = (z == 0) ? Qb : Kb;
    const float* W    = (z == 0) ? Wq : (z == 1) ? Wk : Wv;
    const float* bias = (z == 0) ? bq : (z == 1) ? bk : bvp;
    unsigned short* out = (z == 0) ? qbf : (z == 1) ? kbf : vbf;

    __shared__ unsigned short wldsT[32 * DIM];   // 32 KB

    const int lane = threadIdx.x & 31;
    const int wave = threadIdx.x >> 5;
    const int m0 = blockIdx.x * 256 + wave * 32;
    const int n0 = blockIdx.y * 32;
    const int half = lane >> 4;
    const int l16  = lane & 15;

    floatx8 acc[4];
    #pragma unroll
    for (int t = 0; t < 4; ++t) acc[t] = (floatx8)0.f;
    gemm_core(A, W, wldsT, m0, n0, lane, acc);

    #pragma unroll
    for (int ar = 0; ar < 2; ++ar) {
        #pragma unroll
        for (int t = 0; t < 2; ++t) {
            const floatx8& a = acc[ar * 2 + t];
            const int col = n0 + t * 16 + l16;
            const float bb = bias[col];
            const int h = col >> 6, d = col & 63;
            #pragma unroll
            for (int i = 0; i < 8; ++i) {
                int r  = m0 + ar * 16 + i + 8 * half;     // token row 0..8191
                int bt = r >> 10, nn = r & 1023;
                out[(((size_t)bt * HEADS + h) * NSEQ + nn) * DHEAD + d] = f2bf(a[i] + bb);
            }
        }
    }
}

// ============================================================================
// Kernel 2: flash attention per (b,h). grid = 512, block = 256.
// Wave w handles 16 query rows; chunk = 32 keys staged in LDS; next chunk
// prefetched (global_prefetch_b8) while softmax VALU runs.
// o = q + softmax(q k^T / sqrt(512), mask) v ; written to [B,NQ,512] f32.
// ============================================================================
__global__ __launch_bounds__(256)
void attention_kernel(const unsigned short* __restrict__ qbf,
                      const unsigned short* __restrict__ kbf,
                      const unsigned short* __restrict__ vbf,
                      const int* __restrict__ mask,
                      float* __restrict__ attn_out)
{
    __shared__ unsigned short k_lds[32 * 64];      // 32 keys x 64 (bf16)
    __shared__ unsigned short vT[64 * 40];         // [d][key], padded stride 40
    __shared__ unsigned short pst[8 * 16 * 32];    // per-wave P staging

    const int qblk = blockIdx.x & 7;
    const int bh   = blockIdx.x >> 3;
    const int b    = bh >> 3;
    const int h    = bh & 7;
    const int lane = threadIdx.x & 31;
    const int wave = threadIdx.x >> 5;
    const int half = lane >> 4;
    const int l16  = lane & 15;
    const int m0   = qblk * 128 + wave * 16;
    const int ac = half ? 8 : 0;
    const int bc = half ? 8 : 0;

    // q A-fragments for d = 0..31 and 32..63
    const unsigned short* qrow = qbf + ((size_t)bh * NSEQ + m0 + l16) * DHEAD;
    Frag aq0, aq1;
    aq0.q[0] = *(const uint4*)(qrow + ac);
    aq0.q[1] = *(const uint4*)(qrow + ac + 16);
    aq1.q[0] = *(const uint4*)(qrow + 32 + ac);
    aq1.q[1] = *(const uint4*)(qrow + 32 + ac + 16);

    float mrun[8], lrun[8];
    floatx8 accO[4];
    #pragma unroll
    for (int i = 0; i < 8; ++i) { mrun[i] = -1e30f; lrun[i] = 0.f; }
    #pragma unroll
    for (int t = 0; t < 4; ++t) accO[t] = (floatx8)0.f;

    const float scale = 0.044194173824159216f;   // 1/sqrt(512)
    const int tl = threadIdx.x;

    for (int kc = 0; kc < NSEQ; kc += 32) {
        { // cooperative stage: K chunk row-major, V chunk transposed
            int r  = tl >> 3;          // 0..31 (key within chunk)
            int c8 = (tl & 7) * 8;     // 0..56 (d offset)
            const size_t base = ((size_t)bh * NSEQ + kc + r) * DHEAD + c8;
            *(uint4*)&k_lds[r * 64 + c8] = *(const uint4*)(kbf + base);
            uint4 vv = *(const uint4*)(vbf + base);
            if (kc + 32 < NSEQ) {      // prefetch next chunk into cache
                __builtin_prefetch(kbf + base + 32 * DHEAD, 0, 0);
                __builtin_prefetch(vbf + base + 32 * DHEAD, 0, 0);
            }
            const unsigned short* ve = (const unsigned short*)&vv;
            #pragma unroll
            for (int j = 0; j < 8; ++j) vT[(c8 + j) * 40 + r] = ve[j];
        }
        __syncthreads();

        // S = q k^T  (two 16x16 tiles: keys 0..15 and 16..31)
        floatx8 s[2];
        #pragma unroll
        for (int jb = 0; jb < 2; ++jb) {
            floatx8 sa = (floatx8)0.f;
            Frag bk0, bk1;
            const unsigned short* kb = &k_lds[(jb * 16 + l16) * 64];
            bk0.q[0] = *(const uint4*)(kb + bc);
            bk0.q[1] = *(const uint4*)(kb + bc + 16);
            bk1.q[0] = *(const uint4*)(kb + 32 + bc);
            bk1.q[1] = *(const uint4*)(kb + 32 + bc + 16);
            sa = wmma_bf16(aq0, bk0, sa);
            sa = wmma_bf16(aq1, bk1, sa);
            s[jb] = sa;
        }

        const int mv0 = mask[b * NSEQ + kc + l16];
        const int mv1 = mask[b * NSEQ + kc + 16 + l16];

        float p0[8], p1[8];
        #pragma unroll
        for (int i = 0; i < 8; ++i) {
            float x0 = mv0 ? s[0][i] * scale : -10000.f;
            float x1 = mv1 ? s[1][i] * scale : -10000.f;
            float cmax = fmaxf(x0, x1);
            #pragma unroll
            for (int mm = 1; mm < 16; mm <<= 1)
                cmax = fmaxf(cmax, __shfl_xor(cmax, mm, 32));
            float mnew = fmaxf(mrun[i], cmax);
            float corr = __expf(mrun[i] - mnew);
            mrun[i] = mnew;
            p0[i] = __expf(x0 - mnew);
            p1[i] = __expf(x1 - mnew);
            float rs = p0[i] + p1[i];
            #pragma unroll
            for (int mm = 1; mm < 16; mm <<= 1)
                rs += __shfl_xor(rs, mm, 32);
            lrun[i] = lrun[i] * corr + rs;
            #pragma unroll
            for (int t = 0; t < 4; ++t) accO[t][i] *= corr;
        }

        // C/D layout -> A-frag layout via per-wave LDS staging (no barrier:
        // same-wave DS ops are in program order, compiler inserts dscnt waits)
        unsigned short* pw = &pst[wave * 512];
        #pragma unroll
        for (int i = 0; i < 8; ++i) {
            int rr = i + 8 * half;
            pw[rr * 32 + l16]      = f2bf(p0[i]);
            pw[rr * 32 + 16 + l16] = f2bf(p1[i]);
        }
        Frag apf;
        apf.q[0] = *(const uint4*)&pw[l16 * 32 + ac];
        apf.q[1] = *(const uint4*)&pw[l16 * 32 + ac + 16];

        #pragma unroll
        for (int t = 0; t < 4; ++t) {
            Frag bvf;
            const unsigned short* vp = &vT[(t * 16 + l16) * 40 + bc];
            bvf.q[0] = *(const uint4*)(vp);
            bvf.q[1] = *(const uint4*)(vp + 16);
            accO[t] = wmma_bf16(apf, bvf, accO[t]);
        }
        __syncthreads();
    }

    // epilogue: o = q + accO/l  ->  [B, NQ, 512] f32
    #pragma unroll
    for (int t = 0; t < 4; ++t) {
        int d = t * 16 + l16;
        #pragma unroll
        for (int i = 0; i < 8; ++i) {
            int r = m0 + i + 8 * half;
            float qv = bf2f(qbf[((size_t)bh * NSEQ + r) * DHEAD + d]);
            attn_out[((size_t)(b * NSEQ + r)) * DIM + h * DHEAD + d] =
                qv + accO[t][i] / lrun[i];
        }
    }
}

// ============================================================================
// Kernel 3/5: layernorm over 512 features, one wave per row. grid = 1024.
// Optionally also emits bf16 copy for the following WMMA GEMM.
// ============================================================================
__global__ __launch_bounds__(256)
void layernorm_kernel(const float* __restrict__ in, const float* __restrict__ g,
                      const float* __restrict__ be, float* __restrict__ out_f32,
                      unsigned short* __restrict__ out_bf)
{
    const int lane = threadIdx.x & 31;
    const int wave = threadIdx.x >> 5;
    const size_t row = (size_t)blockIdx.x * 8 + wave;
    const float* x = in + row * DIM;

    float xr[16];
    float s = 0.f, s2 = 0.f;
    #pragma unroll
    for (int j = 0; j < 16; ++j) {
        float v = x[lane + j * 32];
        xr[j] = v; s += v; s2 += v * v;
    }
    #pragma unroll
    for (int mm = 1; mm < 32; mm <<= 1) {
        s  += __shfl_xor(s,  mm, 32);
        s2 += __shfl_xor(s2, mm, 32);
    }
    const float mu  = s * (1.f / DIM);
    const float var = s2 * (1.f / DIM) - mu * mu;
    const float rstd = rsqrtf(var + 1e-5f);
    #pragma unroll
    for (int j = 0; j < 16; ++j) {
        int c = lane + j * 32;
        float y = (xr[j] - mu) * rstd * g[c] + be[c];
        out_f32[row * DIM + c] = y;
        if (out_bf) out_bf[row * DIM + c] = f2bf(y);
    }
}

// ============================================================================
// Kernel 4: y = x + relu(x @ Wo + bo). grid = (32, 16). A = bf16 LN output.
// ============================================================================
__global__ __launch_bounds__(256)
void out_proj_kernel(const unsigned short* __restrict__ xbf,
                     const float* __restrict__ xf,
                     const float* __restrict__ Wo, const float* __restrict__ bo,
                     float* __restrict__ y)
{
    __shared__ unsigned short wldsT[32 * DIM];   // 32 KB

    const int lane = threadIdx.x & 31;
    const int wave = threadIdx.x >> 5;
    const int m0 = blockIdx.x * 256 + wave * 32;
    const int n0 = blockIdx.y * 32;
    const int half = lane >> 4;
    const int l16  = lane & 15;

    floatx8 acc[4];
    #pragma unroll
    for (int t = 0; t < 4; ++t) acc[t] = (floatx8)0.f;
    gemm_core(xbf, Wo, wldsT, m0, n0, lane, acc);

    #pragma unroll
    for (int ar = 0; ar < 2; ++ar) {
        #pragma unroll
        for (int t = 0; t < 2; ++t) {
            const floatx8& a = acc[ar * 2 + t];
            const int col = n0 + t * 16 + l16;
            const float bb = bo[col];
            #pragma unroll
            for (int i = 0; i < 8; ++i) {
                int r = m0 + ar * 16 + i + 8 * half;
                float v = xf[(size_t)r * DIM + col] + fmaxf(a[i] + bb, 0.f);
                y[(size_t)r * DIM + col] = v;
            }
        }
    }
}

// ============================================================================
extern "C" void kernel_launch(void* const* d_in, const int* in_sizes, int n_in,
                              void* d_out, int out_size, void* d_ws, size_t ws_size,
                              hipStream_t stream)
{
    const float* Q   = (const float*)d_in[0];
    const float* K   = (const float*)d_in[1];
    const int*   msk = (const int*)  d_in[2];
    const float* Wq  = (const float*)d_in[3];
    const float* bq  = (const float*)d_in[4];
    const float* Wk  = (const float*)d_in[5];
    const float* bk  = (const float*)d_in[6];
    const float* Wv  = (const float*)d_in[7];
    const float* bv  = (const float*)d_in[8];
    const float* Wo  = (const float*)d_in[9];
    const float* bo  = (const float*)d_in[10];
    const float* g0  = (const float*)d_in[11];
    const float* b0  = (const float*)d_in[12];
    const float* g1  = (const float*)d_in[13];
    const float* b1  = (const float*)d_in[14];

    const size_t NELEM = (size_t)BATCH * NSEQ * DIM;   // 4,194,304
    char* ws = (char*)d_ws;
    unsigned short* Qb  = (unsigned short*)ws;  ws += NELEM * 2;
    unsigned short* Kb  = (unsigned short*)ws;  ws += NELEM * 2;
    unsigned short* qbf = (unsigned short*)ws;  ws += NELEM * 2;
    unsigned short* kbf = (unsigned short*)ws;  ws += NELEM * 2;
    unsigned short* vbf = (unsigned short*)ws;  ws += NELEM * 2;
    float* attn = (float*)ws;                   ws += NELEM * 4;
    float* xf   = (float*)ws;                   ws += NELEM * 4;
    float* yb   = (float*)ws;                   ws += NELEM * 4;
    unsigned short* xbf = (unsigned short*)ws;  ws += NELEM * 2;

    dim3 blk(256);
    cvt_bf16_kernel<<<dim3(2048, 2), blk, 0, stream>>>(Q, K, Qb, Kb);
    qkv_proj_kernel<<<dim3(32, 16, 3), blk, 0, stream>>>(Qb, Kb, Wq, bq, Wk, bk, Wv, bv,
                                                         qbf, kbf, vbf);
    attention_kernel<<<dim3(512), blk, 0, stream>>>(qbf, kbf, vbf, msk, attn);
    layernorm_kernel<<<dim3(1024), blk, 0, stream>>>(attn, g0, b0, xf, xbf);
    out_proj_kernel<<<dim3(32, 16), blk, 0, stream>>>(xbf, xf, Wo, bo, yb);
    layernorm_kernel<<<dim3(1024), blk, 0, stream>>>(yb, g1, b1, (float*)d_out, nullptr);
}
